// Kalman_Filter_17446157156408
// MI455X (gfx1250) — compile-verified, matrix operand order checked by reading, not run on hardware
//
#include <hip/hip_runtime.h>
#include <cstdint>
#include <cstddef>

#define BSn 1024
#define Tn  128
#define DAn 8
#define DZn 16
#define Kn  3
#define Hn  128
#define G4n 512   // 4*H
#define RB  16    // batch rows per block (LSTM kernels)

typedef __attribute__((ext_vector_type(16))) __bf16 v16bf;
typedef __attribute__((ext_vector_type(8)))  float  v8f;

union FragBF { v16bf v; unsigned int u[8]; };

__device__ __forceinline__ unsigned short f2bf(float f) {
  unsigned int u = __float_as_uint(f);
  u += 0x7FFFu + ((u >> 16) & 1u);   // round-to-nearest-even
  return (unsigned short)(u >> 16);
}
__device__ __forceinline__ float bf2f(unsigned short u) {
  return __uint_as_float(((unsigned int)u) << 16);
}
__device__ __forceinline__ float sigm(float x) { return 1.0f / (1.0f + __expf(-x)); }

#if __has_builtin(__builtin_amdgcn_tensor_load_to_lds)
#define HAS_TDM 1
#else
#define HAS_TDM 0
#endif

#if HAS_TDM
typedef unsigned int v4u __attribute__((ext_vector_type(4)));
typedef int v8i __attribute__((ext_vector_type(8)));
typedef int v4i __attribute__((ext_vector_type(4)));

// 2D bf16 tile (rows x cols, contiguous) global -> LDS via Tensor Data Mover.
// 6-arg builtin form: (g0, g1, g2, g3, g_extra, cpol)
__device__ __forceinline__ void tdm_load_2d_bf16(unsigned lds_addr,
                                                 const unsigned short* gptr,
                                                 unsigned rows, unsigned cols) {
  unsigned long long ga = (unsigned long long)(uintptr_t)gptr;
  v4u g0;
  g0[0] = 1u;                                    // count=1, user descriptor
  g0[1] = lds_addr;                              // LDS byte address
  g0[2] = (unsigned)(ga & 0xFFFFFFFFu);          // global_addr[31:0]
  g0[3] = (unsigned)((ga >> 32) & 0x1FFFFFFu)    // global_addr[56:32]
        | (2u << 30);                            // type=2 ("image")
  v8i g1;
  g1[0] = (int)(1u << 16);                       // data_size=1 -> 2 bytes
  g1[1] = (int)((cols & 0xFFFFu) << 16);         // tensor_dim0[15:0]
  g1[2] = (int)(((cols >> 16) & 0xFFFFu) |       // tensor_dim0[31:16]
                ((rows & 0xFFFFu) << 16));       // tensor_dim1[15:0]
  g1[3] = (int)(((rows >> 16) & 0xFFFFu) |       // tensor_dim1[31:16]
                ((cols & 0xFFFFu) << 16));       // tile_dim0
  g1[4] = (int)(rows & 0xFFFFu);                 // tile_dim1 (tile_dim2=0)
  g1[5] = (int)cols;                             // tensor_dim0_stride[31:0]
  g1[6] = 0;
  g1[7] = 0;
  v4i z4 = {0, 0, 0, 0};
  v8i z8 = {0, 0, 0, 0, 0, 0, 0, 0};
  __builtin_amdgcn_tensor_load_to_lds(g0, g1, z4, z4, z8, 0);
}
#endif

// ---------------------------------------------------------------------------
// Prep: fp32 -> bf16 weight conversion (Whh0, Wih1, Whh1).
// ---------------------------------------------------------------------------
__global__ void prep_kernel(const float* __restrict__ Whh0,
                            const float* __restrict__ Wih1,
                            const float* __restrict__ Whh1,
                            unsigned short* whh0_b, unsigned short* wih1_b,
                            unsigned short* whh1_b) {
  int idx = blockIdx.x * blockDim.x + threadIdx.x;
  if (idx < G4n * Hn) {
    whh0_b[idx] = f2bf(Whh0[idx]);
    wih1_b[idx] = f2bf(Wih1[idx]);
    whh1_b[idx] = f2bf(Whh1[idx]);
  }
}

// ---------------------------------------------------------------------------
// LSTM layer 0, full time loop per block. Block = 16 batch rows, 256 threads.
// LDS: Whh0 (128KB, TDM) | gates 16x512 | h 16x128 bf16 | x 16x8 | bias 512 |
//      Wih0 512x8
// ---------------------------------------------------------------------------
__global__ void lstm_layer0(const unsigned short* __restrict__ whh0_b, // [512,128] bf16
                            const float* __restrict__ Wih0,            // [512,8]
                            const float* __restrict__ bih,
                            const float* __restrict__ bhh,
                            const float* __restrict__ a,               // [B,T,8]
                            const float* __restrict__ a0,              // [8]
                            unsigned short* __restrict__ h0_all) {     // [B,T,128] bf16
  extern __shared__ char smem[];
  unsigned short* sW  = (unsigned short*)smem;                    // 131072 B
  float*          sG  = (float*)(smem + 131072);                  // 32768 B
  unsigned short* sH  = (unsigned short*)(smem + 163840);         // 4096 B
  float*          sX  = (float*)(smem + 167936);                  // 512 B
  float*          sB  = (float*)(smem + 168448);                  // 2048 B
  float*          sWi = (float*)(smem + 170496);                  // 16384 B

  const int tid  = threadIdx.x;
  const int wv   = tid >> 5;
  const int lane = tid & 31;
  const int m    = lane & 15;
  const int half = lane >> 4;
  const int b0g  = blockIdx.x * RB;
  const int n0   = wv * 64;

  // stage weights into LDS
#if HAS_TDM
  if (tid < 32) {
    tdm_load_2d_bf16((unsigned)(uintptr_t)sW, whh0_b, G4n, Hn);
    __builtin_amdgcn_s_wait_tensorcnt(0);
  }
#else
  for (int i = tid * 8; i < G4n * Hn; i += 256 * 8)
    *(uint4*)(sW + i) = *(const uint4*)(whh0_b + i);
#endif
  // stage t-invariant epilogue constants; zero h state
  for (int i = tid; i < G4n; i += 256) sB[i] = bih[i] + bhh[i];
  for (int i = tid; i < G4n * DAn; i += 256) sWi[i] = Wih0[i];
  for (int i = tid; i < RB * Hn; i += 256) sH[i] = 0;
  __syncthreads();

  float creg[8];
  #pragma unroll
  for (int e = 0; e < 8; ++e) creg[e] = 0.0f;

  for (int t = 0; t < Tn; ++t) {
    // stage x_t tile (16 x 8)
    if (tid < RB * DAn) {
      int b = tid >> 3, k = tid & 7;
      sX[tid] = (t == 0) ? a0[k]
                         : a[((size_t)(b0g + b) * Tn + (t - 1)) * DAn + k];
    }
    __syncthreads();

    // gates GEMM: 16 x 64 per wave, K=128
    v8f acc[4] = {};
    #pragma unroll
    for (int s = 0; s < 4; ++s) {
      FragBF af;
      #pragma unroll
      for (int v = 0; v < 8; ++v) {
        int k = s * 32 + ((v >> 2) << 4) + (half << 3) + ((v & 3) << 1);
        af.u[v] = *(const unsigned int*)(sH + m * Hn + k);
      }
      #pragma unroll
      for (int nt = 0; nt < 4; ++nt) {
        FragBF bf;
        #pragma unroll
        for (int v = 0; v < 8; ++v) {
          int k = s * 32 + ((v >> 2) << 4) + (half << 3) + ((v & 3) << 1);
          bf.u[v] = *(const unsigned int*)(sW + (n0 + nt * 16 + m) * Hn + k);
        }
        acc[nt] = __builtin_amdgcn_wmma_f32_16x16x32_bf16(false, af.v, false, bf.v,
                                                          (short)0, acc[nt], false, false);
      }
    }
    // epilogue: + bias + Wih0 . x_t (constants from LDS) -> gates LDS
    #pragma unroll
    for (int nt = 0; nt < 4; ++nt) {
      int j = n0 + nt * 16 + m;
      float bias = sB[j];
      float w0 = sWi[j * DAn + 0], w1 = sWi[j * DAn + 1];
      float w2 = sWi[j * DAn + 2], w3 = sWi[j * DAn + 3];
      float w4 = sWi[j * DAn + 4], w5 = sWi[j * DAn + 5];
      float w6 = sWi[j * DAn + 6], w7 = sWi[j * DAn + 7];
      #pragma unroll
      for (int v = 0; v < 8; ++v) {
        int brow = v + 8 * half;
        const float* x = sX + brow * DAn;
        float s = acc[nt][v] + bias;
        s += w0 * x[0] + w1 * x[1] + w2 * x[2] + w3 * x[3];
        s += w4 * x[4] + w5 * x[5] + w6 * x[6] + w7 * x[7];
        sG[brow * G4n + j] = s;
      }
    }
    __syncthreads();

    // pointwise: 8 elements per thread
    #pragma unroll
    for (int e = 0; e < 8; ++e) {
      int idx = tid + 256 * e;
      int b = idx >> 7, j = idx & 127;
      float gi = sG[b * G4n + j];
      float gf = sG[b * G4n + Hn + j];
      float gg = sG[b * G4n + 2 * Hn + j];
      float go = sG[b * G4n + 3 * Hn + j];
      float cn = sigm(gf) * creg[e] + sigm(gi) * tanhf(gg);
      float hn = sigm(go) * tanhf(cn);
      creg[e] = cn;
      unsigned short hb = f2bf(hn);
      sH[b * Hn + j] = hb;
      h0_all[((size_t)(b0g + b) * Tn + t) * Hn + j] = hb;
    }
    __syncthreads();
  }
}

// ---------------------------------------------------------------------------
// LSTM layer 1 + linear head + softmax. Block = 16 batch rows, 256 threads.
// LDS: Wih1|Whh1 (256KB, TDM) | gates | h1 | Wlin (3x128) | blin
// ---------------------------------------------------------------------------
__global__ void lstm_layer1(const unsigned short* __restrict__ wih1_b,
                            const unsigned short* __restrict__ whh1_b,
                            const unsigned short* __restrict__ h0_all,  // [B,T,128] bf16
                            const float* __restrict__ bih,
                            const float* __restrict__ bhh,
                            const float* __restrict__ Wlin,
                            const float* __restrict__ blin,
                            float* __restrict__ alpha_out) {            // [B,T,3]
  extern __shared__ char smem[];
  unsigned short* sWi = (unsigned short*)smem;                        // 131072
  unsigned short* sWh = (unsigned short*)(smem + 131072);             // 131072
  float*          sG  = (float*)(smem + 262144);                      // 32768
  unsigned short* sH  = (unsigned short*)(smem + 294912);             // 4096
  float*          sWl = (float*)(smem + 299008);                      // 1536
  float*          sBl = (float*)(smem + 300544);                      // 16
  float*          sB  = (float*)(smem + 300560);                      // 2048

  const int tid  = threadIdx.x;
  const int wv   = tid >> 5;
  const int lane = tid & 31;
  const int m    = lane & 15;
  const int half = lane >> 4;
  const int b0g  = blockIdx.x * RB;
  const int n0   = wv * 64;

#if HAS_TDM
  if (tid < 32) {
    tdm_load_2d_bf16((unsigned)(uintptr_t)sWi, wih1_b, G4n, Hn);
    tdm_load_2d_bf16((unsigned)(uintptr_t)sWh, whh1_b, G4n, Hn);
    __builtin_amdgcn_s_wait_tensorcnt(0);
  }
#else
  for (int i = tid * 8; i < G4n * Hn; i += 256 * 8) {
    *(uint4*)(sWi + i) = *(const uint4*)(wih1_b + i);
    *(uint4*)(sWh + i) = *(const uint4*)(whh1_b + i);
  }
#endif
  for (int i = tid; i < RB * Hn; i += 256) sH[i] = 0;
  for (int i = tid; i < Kn * Hn; i += 256) sWl[i] = Wlin[i];
  for (int i = tid; i < G4n; i += 256) sB[i] = bih[i] + bhh[i];
  if (tid < Kn) sBl[tid] = blin[tid];
  __syncthreads();

  float creg[8];
  #pragma unroll
  for (int e = 0; e < 8; ++e) creg[e] = 0.0f;

  for (int t = 0; t < Tn; ++t) {
    __syncthreads();
    v8f acc[4] = {};
    // chain 1: x = h0_all[b, t, :]  (global, bf16) with Wih1
    #pragma unroll
    for (int s = 0; s < 4; ++s) {
      FragBF af;
      #pragma unroll
      for (int v = 0; v < 8; ++v) {
        int k = s * 32 + ((v >> 2) << 4) + (half << 3) + ((v & 3) << 1);
        af.u[v] = *(const unsigned int*)(h0_all + ((size_t)(b0g + m) * Tn + t) * Hn + k);
      }
      #pragma unroll
      for (int nt = 0; nt < 4; ++nt) {
        FragBF bf;
        #pragma unroll
        for (int v = 0; v < 8; ++v) {
          int k = s * 32 + ((v >> 2) << 4) + (half << 3) + ((v & 3) << 1);
          bf.u[v] = *(const unsigned int*)(sWi + (n0 + nt * 16 + m) * Hn + k);
        }
        acc[nt] = __builtin_amdgcn_wmma_f32_16x16x32_bf16(false, af.v, false, bf.v,
                                                          (short)0, acc[nt], false, false);
      }
    }
    // chain 2: recurrent h1 (LDS) with Whh1
    #pragma unroll
    for (int s = 0; s < 4; ++s) {
      FragBF af;
      #pragma unroll
      for (int v = 0; v < 8; ++v) {
        int k = s * 32 + ((v >> 2) << 4) + (half << 3) + ((v & 3) << 1);
        af.u[v] = *(const unsigned int*)(sH + m * Hn + k);
      }
      #pragma unroll
      for (int nt = 0; nt < 4; ++nt) {
        FragBF bf;
        #pragma unroll
        for (int v = 0; v < 8; ++v) {
          int k = s * 32 + ((v >> 2) << 4) + (half << 3) + ((v & 3) << 1);
          bf.u[v] = *(const unsigned int*)(sWh + (n0 + nt * 16 + m) * Hn + k);
        }
        acc[nt] = __builtin_amdgcn_wmma_f32_16x16x32_bf16(false, af.v, false, bf.v,
                                                          (short)0, acc[nt], false, false);
      }
    }
    #pragma unroll
    for (int nt = 0; nt < 4; ++nt) {
      int j = n0 + nt * 16 + m;
      float bias = sB[j];
      #pragma unroll
      for (int v = 0; v < 8; ++v) {
        int brow = v + 8 * half;
        sG[brow * G4n + j] = acc[nt][v] + bias;
      }
    }
    __syncthreads();

    #pragma unroll
    for (int e = 0; e < 8; ++e) {
      int idx = tid + 256 * e;
      int b = idx >> 7, j = idx & 127;
      float gi = sG[b * G4n + j];
      float gf = sG[b * G4n + Hn + j];
      float gg = sG[b * G4n + 2 * Hn + j];
      float go = sG[b * G4n + 3 * Hn + j];
      float cn = sigm(gf) * creg[e] + sigm(gi) * tanhf(gg);
      float hn = sigm(go) * tanhf(cn);
      creg[e] = cn;
      sH[b * Hn + j] = f2bf(hn);
    }
    __syncthreads();

    // linear head + softmax, one thread per batch row
    if (tid < RB) {
      int b = tid;
      float lg[Kn];
      #pragma unroll
      for (int kk = 0; kk < Kn; ++kk) {
        float s = sBl[kk];
        for (int j = 0; j < Hn; ++j) s += sWl[kk * Hn + j] * bf2f(sH[b * Hn + j]);
        lg[kk] = s;
      }
      float mx = fmaxf(fmaxf(lg[0], lg[1]), lg[2]);
      float e0 = __expf(lg[0] - mx), e1 = __expf(lg[1] - mx), e2 = __expf(lg[2] - mx);
      float inv = 1.0f / (e0 + e1 + e2);
      float* o = alpha_out + ((size_t)(b0g + b) * Tn + t) * Kn;
      o[0] = e0 * inv; o[1] = e1 * inv; o[2] = e2 * inv;
    }
  }
}

// ---------------------------------------------------------------------------
// Kalman filter: one 256-thread block per batch element, all state in LDS.
// ---------------------------------------------------------------------------
__global__ void kalman_kernel(const float* __restrict__ a,
                              const float* __restrict__ A_param,   // [3,16,16]
                              const float* __restrict__ C_param,   // [3,8,16]
                              const float* __restrict__ alpha,     // [B,T,3]
                              float* __restrict__ o_muT,  float* __restrict__ o_sigT,
                              float* __restrict__ o_means, float* __restrict__ o_cov,
                              float* __restrict__ o_nm,   float* __restrict__ o_ncov,
                              float* __restrict__ o_A,    float* __restrict__ o_C) {
  const int b = blockIdx.x;
  const int tid = threadIdx.x;
  const int i16 = tid >> 4, j16 = tid & 15;

  __shared__ float sA[256], sAn[256], sC[128], sSig[256], sT1[256], sT2[256];
  __shared__ float sM[128], sX[128], sKg[128], sCS[128];
  __shared__ float sMu[16], sMuP[16], sR[8], sCol[8];

  if (tid < 16) sMuP[tid] = 0.0f;
  sSig[tid] = (i16 == j16) ? 1.0f : 0.0f;
  __syncthreads();

  for (int t = 0; t < Tn; ++t) {
    const size_t bt = (size_t)b * Tn + t;
    const int t1 = (t + 1 < Tn) ? (t + 1) : (Tn - 1);
    const float* al = alpha + bt * Kn;
    const float* an = alpha + ((size_t)b * Tn + t1) * Kn;
    float a0_ = al[0], a1_ = al[1], a2_ = al[2];
    float n0_ = an[0], n1_ = an[1], n2_ = an[2];

    sA[tid]  = a0_ * A_param[tid] + a1_ * A_param[256 + tid] + a2_ * A_param[512 + tid];
    sAn[tid] = n0_ * A_param[tid] + n1_ * A_param[256 + tid] + n2_ * A_param[512 + tid];
    if (tid < 128)
      sC[tid] = a0_ * C_param[tid] + a1_ * C_param[128 + tid] + a2_ * C_param[256 + tid];
    __syncthreads();

    o_A[bt * 256 + tid] = sA[tid];
    if (tid < 128) o_C[bt * 128 + tid] = sC[tid];
    if (tid < 16)  o_nm[bt * 16 + tid] = sMuP[tid];
    o_ncov[bt * 256 + tid] = sSig[tid];

    if (tid < 8) {
      float s = a[bt * DAn + tid];
      for (int k = 0; k < 16; ++k) s -= sC[tid * 16 + k] * sMuP[k];
      sR[tid] = s;
    }
    if (tid < 128) {
      int i = tid >> 4, j = tid & 15;
      float s = 0.0f;
      for (int k = 0; k < 16; ++k) s += sC[i * 16 + k] * sSig[k * 16 + j];
      sCS[tid] = s;
    }
    __syncthreads();

    if (tid < 128) {
      int i = tid >> 4, j = tid & 15;
      if (j < 8) {
        float s = (i == j) ? 1.0f : 0.0f;
        for (int k = 0; k < 16; ++k) s += sCS[i * 16 + k] * sC[j * 16 + k];
        sM[tid] = s;
      } else {
        sM[tid] = ((j - 8) == i) ? 1.0f : 0.0f;
      }
    }
    __syncthreads();

    for (int p = 0; p < 8; ++p) {
      if (tid < 8) sCol[tid] = sM[tid * 16 + p];
      __syncthreads();
      if (tid < 16) sM[p * 16 + tid] = sM[p * 16 + tid] / sCol[p];
      __syncthreads();
      if (tid < 128) {
        int i = tid >> 4, j = tid & 15;
        if (i != p) sM[tid] -= sCol[i] * sM[p * 16 + j];
      }
      __syncthreads();
    }

    if (tid < 128) {
      int i = tid >> 3, j = tid & 7;
      float s = 0.0f;
      for (int k = 0; k < 16; ++k) s += sSig[i * 16 + k] * sC[j * 16 + k];
      sX[tid] = s;
    }
    __syncthreads();
    if (tid < 128) {
      int i = tid >> 3, j = tid & 7;
      float s = 0.0f;
      for (int k = 0; k < 8; ++k) s += sX[i * 8 + k] * sM[k * 16 + 8 + j];
      sKg[tid] = s;
    }
    __syncthreads();

    if (tid < 16) {
      float s = sMuP[tid];
      for (int k = 0; k < 8; ++k) s += sKg[tid * 8 + k] * sR[k];
      sMu[tid] = s;
    }
    {
      float s = (i16 == j16) ? 1.0f : 0.0f;
      for (int k = 0; k < 8; ++k) s -= sKg[i16 * 8 + k] * sC[k * 16 + j16];
      sT1[tid] = s;
    }
    __syncthreads();
    {
      float s = 0.0f;
      for (int k = 0; k < 16; ++k) s += sT1[i16 * 16 + k] * sSig[k * 16 + j16];
      sT2[tid] = s;
    }
    __syncthreads();
    sT1[tid] = 0.5f * (sT2[tid] + sT2[j16 * 16 + i16]);
    __syncthreads();

    if (tid < 16) o_means[bt * 16 + tid] = sMu[tid];
    o_cov[bt * 256 + tid] = sT1[tid];
    if (t == Tn - 1) {
      if (tid < 16) o_muT[(size_t)b * 16 + tid] = sMu[tid];
      o_sigT[(size_t)b * 256 + tid] = sT1[tid];
    }

    if (tid < 16) {
      float s = 0.0f;
      for (int k = 0; k < 16; ++k) s += sAn[tid * 16 + k] * sMu[k];
      sMuP[tid] = s;
    }
    {
      float s = 0.0f;
      for (int k = 0; k < 16; ++k) s += sAn[i16 * 16 + k] * sT1[k * 16 + j16];
      sT2[tid] = s;
    }
    __syncthreads();
    {
      float s = (i16 == j16) ? 1.0f : 0.0f;
      for (int k = 0; k < 16; ++k) s += sT2[i16 * 16 + k] * sAn[j16 * 16 + k];
      sA[tid] = s;
    }
    __syncthreads();
    sSig[tid] = 0.5f * (sA[tid] + sA[j16 * 16 + i16]);
    __syncthreads();
  }
}

// ---------------------------------------------------------------------------
extern "C" void kernel_launch(void* const* d_in, const int* in_sizes, int n_in,
                              void* d_out, int out_size, void* d_ws, size_t ws_size,
                              hipStream_t stream) {
  (void)in_sizes; (void)n_in; (void)out_size; (void)ws_size;

  const float* a       = (const float*)d_in[0];
  const float* A_param = (const float*)d_in[1];
  const float* C_param = (const float*)d_in[2];
  const float* a0      = (const float*)d_in[3];
  const float* Wih0    = (const float*)d_in[4];
  const float* Whh0    = (const float*)d_in[5];
  const float* bih0    = (const float*)d_in[6];
  const float* bhh0    = (const float*)d_in[7];
  const float* Wih1    = (const float*)d_in[8];
  const float* Whh1    = (const float*)d_in[9];
  const float* bih1    = (const float*)d_in[10];
  const float* bhh1    = (const float*)d_in[11];
  const float* Wlin    = (const float*)d_in[12];
  const float* blin    = (const float*)d_in[13];

  char* ws = (char*)d_ws;
  unsigned short* whh0_b = (unsigned short*)(ws + 0);        // 131072 B
  unsigned short* wih1_b = (unsigned short*)(ws + 131072);   // 131072 B
  unsigned short* whh1_b = (unsigned short*)(ws + 262144);   // 131072 B
  unsigned short* h0_all = (unsigned short*)(ws + 393216);   // [B,T,128] bf16, 32 MB

  float* out = (float*)d_out;
  float* o_muT   = out;               // [B,16]
  float* o_sigT  = out + 16384;       // [B,16,16]
  float* o_means = out + 278528;      // [B,T,16]
  float* o_cov   = out + 2375680;     // [B,T,16,16]
  float* o_nm    = out + 35930112;    // [B,T,16]
  float* o_ncov  = out + 38027264;    // [B,T,16,16]
  float* o_A     = out + 71581696;    // [B,T,16,16]
  float* o_C     = out + 105136128;   // [B,T,8,16]
  float* o_alpha = out + 121913344;   // [B,T,3]

  constexpr size_t L0_SMEM = 131072 + 32768 + 4096 + 512 + 2048 + 16384;
  constexpr size_t L1_SMEM = 262144 + 32768 + 4096 + 1536 + 16 + 2048;

  prep_kernel<<<256, 256, 0, stream>>>(Whh0, Wih1, Whh1, whh0_b, wih1_b, whh1_b);

  lstm_layer0<<<BSn / RB, 256, L0_SMEM, stream>>>(whh0_b, Wih0, bih0, bhh0,
                                                  a, a0, h0_all);

  lstm_layer1<<<BSn / RB, 256, L1_SMEM, stream>>>(wih1_b, whh1_b, h0_all,
                                                  bih1, bhh1, Wlin, blin, o_alpha);

  kalman_kernel<<<BSn, 256, 0, stream>>>(a, A_param, C_param, o_alpha,
                                         o_muT, o_sigT, o_means, o_cov,
                                         o_nm, o_ncov, o_A, o_C);
}
